// LSTMModel_5239860101211
// MI455X (gfx1250) — compile-verified, matrix operand order checked by reading, not run on hardware
//
#include <hip/hip_runtime.h>
#include <cstddef>

typedef __attribute__((ext_vector_type(16))) _Float16 v16h;
typedef __attribute__((ext_vector_type(8)))  _Float16 v8h;
typedef __attribute__((ext_vector_type(8)))  float    v8f;
typedef __attribute__((ext_vector_type(4)))  float    v4f;
typedef __attribute__((ext_vector_type(2)))  float    v2f;

#define HSZ   64
#define TLEN  1024
#define BROWS 16        // batch rows per workgroup

// fast sigmoid: one v_exp_f32 + one v_rcp_f32
__device__ __forceinline__ float fast_sigmoid(float v) {
    return __builtin_amdgcn_rcpf(1.0f + __expf(-v));
}
// fast tanh: tanh(x) = 1 - 2/(1 + exp(2x)); saturates correctly via exp over/underflow
__device__ __forceinline__ float fast_tanh(float v) {
    return fmaf(-2.0f, __builtin_amdgcn_rcpf(1.0f + __expf(2.0f * v)), 1.0f);
}

// Persistent fused LSTM + FC kernel.
// grid = B/16 = 64 workgroups, block = 128 threads (4 waves of 32).
// Wave w computes gate w (i,f,g,o) for its WG's 16 batch rows:
//   G[16, 64] = bias + X[16,4] @ W_ih_w^T  (v_wmma_f32_16x16x4_f32, pure fp32)
//             + H[16,64] @ W_hh_w^T        (2x v_wmma_f32_16x16x32_f16, f32 acc)
// Gates exchanged through LDS; c kept in registers; h kept in LDS as f16.
__global__ __launch_bounds__(128, 1)
void lstm_persistent(const float* __restrict__ x,
                     const float* __restrict__ W_ih,
                     const float* __restrict__ W_hh,
                     const float* __restrict__ b_ih,
                     const float* __restrict__ b_hh,
                     const float* __restrict__ W_fc,
                     const float* __restrict__ b_fc,
                     float* __restrict__ out)
{
    __shared__ float    s_g[4 * BROWS * HSZ];   // activated gates [gate][row][unit], 16 KB
    __shared__ _Float16 s_h[BROWS * HSZ];       // h state, f16 row-major, 2 KB

    const int tid  = threadIdx.x;
    const int wv   = tid >> 5;          // wave id == gate id (i,f,g,o)
    const int lane = tid & 31;
    const int p    = lane & 15;         // M (A/C) or N (B/C) index
    const int hi   = lane >> 4;         // half-wave selector in fragment layouts
    const int bb   = blockIdx.x * BROWS;

    // Branchless gate activation: act = fma(ka, sigmoid(ka*v), kb)
    //   sigmoid gates (i,f,o): ka=1, kb=0
    //   tanh gate (g):         ka=2, kb=-1   (tanh(v) = 2*sigmoid(2v)-1)
    const float ka = (wv == 2) ? 2.0f : 1.0f;   // wave-uniform
    const float kb = (wv == 2) ? -1.0f : 0.0f;

    // ---------------- prologue: resident weight fragments ----------------
    // W_hh B-fragments (f16): 4 N-tiles x 2 K-chunks. col = 64*wv + 16*n + p.
    // Element e of a 16-bit fragment holds K = kc*32 + (e<8 ? e : e+8) + 8*hi.
    v16h bh[4][2];
    v2f  bx[4];          // W_ih B-fragments (f32, K=4): element e -> K = e + 2*hi
    float bs[4];         // bias (b_ih + b_hh) per N-column
#pragma unroll
    for (int n = 0; n < 4; ++n) {
        const int col = 64 * wv + 16 * n + p;
#pragma unroll
        for (int kc = 0; kc < 2; ++kc) {
#pragma unroll
            for (int e = 0; e < 16; ++e) {
                const int k = kc * 32 + ((e < 8) ? e : e + 8) + 8 * hi;
                bh[n][kc][e] = (_Float16)W_hh[col * HSZ + k];
            }
        }
        bx[n][0] = W_ih[col * 4 + 2 * hi];
        bx[n][1] = W_ih[col * 4 + 2 * hi + 1];
        bs[n]    = b_ih[col] + b_hh[col];
    }

    // Elementwise-phase lane mapping: 128 lanes x 8 elements = 16 rows x 64 units.
    const int erow = tid >> 3;          // row 0..15 (8 lanes per row -> shfl reduce)
    const int eu0  = (tid & 7) * 8;     // first of 8 consecutive hidden units
    float wfc[8];
#pragma unroll
    for (int j = 0; j < 8; ++j) wfc[j] = W_fc[eu0 + j];
    const float bfc = b_fc[0];

    float c8[8];
#pragma unroll
    for (int j = 0; j < 8; ++j) c8[j] = 0.0f;

    // zero h state (128 lanes x 8 halfs = 1024 elements)
    {
        v8h z;
#pragma unroll
        for (int j = 0; j < 8; ++j) z[j] = (_Float16)0.0f;
        ((v8h*)s_h)[tid] = z;
    }
    __syncthreads();

    // x A-fragment for t=0 (software-pipelined global load):
    // lane holds x[row, t, 2*hi .. 2*hi+1]
    const float* xrow = x + (size_t)(bb + p) * TLEN * 4 + 2 * hi;
    v2f ax;
    ax[0] = xrow[0];
    ax[1] = xrow[1];

    // ---------------- serial recurrence over T ----------------
    for (int t = 0; t < TLEN; ++t) {
        // A-fragments of h (f16) from LDS: two contiguous 16B runs per fragment.
        const v8h* hv = (const v8h*)(&s_h[p * HSZ]);
        const v8h l0 = hv[0 + hi];   // K =      8*hi .. +7
        const v8h l1 = hv[2 + hi];   // K = 16 + 8*hi .. +7
        const v8h l2 = hv[4 + hi];   // K = 32 + 8*hi .. +7
        const v8h l3 = hv[6 + hi];   // K = 48 + 8*hi .. +7
        v16h a0, a1;
#pragma unroll
        for (int e = 0; e < 8; ++e) {
            a0[e] = l0[e]; a0[e + 8] = l1[e];
            a1[e] = l2[e]; a1[e + 8] = l3[e];
        }

        // issue next step's x load now; consumed next iteration (latency hidden
        // under the WMMAs, activations and barrier of this step)
        const int tn = (t + 1 < TLEN) ? (t + 1) : t;
        v2f ax_next;
        ax_next[0] = xrow[tn * 4];
        ax_next[1] = xrow[tn * 4 + 1];

#pragma unroll
        for (int n = 0; n < 4; ++n) {
            v8f acc;
#pragma unroll
            for (int r = 0; r < 8; ++r) acc[r] = bs[n];
            // pure-fp32 input projection (K=4)
            acc = __builtin_amdgcn_wmma_f32_16x16x4_f32(
                      false, ax, false, bx[n], (short)0, acc, false, false);
            // recurrent matmul, f16 operands / f32 accumulate (K=64)
            acc = __builtin_amdgcn_wmma_f32_16x16x32_f16(
                      false, a0, false, bh[n][0], (short)0, acc, false, false);
            acc = __builtin_amdgcn_wmma_f32_16x16x32_f16(
                      false, a1, false, bh[n][1], (short)0, acc, false, false);

            // branchless activation, publish to LDS
#pragma unroll
            for (int r = 0; r < 8; ++r) {
                const float s   = fast_sigmoid(ka * acc[r]);
                const float act = fmaf(ka, s, kb);
                s_g[(wv * BROWS + (r + 8 * hi)) * HSZ + 16 * n + p] = act;
            }
        }
        ax = ax_next;
        __syncthreads();

        // ------- elementwise c/h update + fused FC dot -------
        const int gbase = erow * HSZ + eu0;
        const v4f iA = *(const v4f*)&s_g[0 * 1024 + gbase];
        const v4f iB = *(const v4f*)&s_g[0 * 1024 + gbase + 4];
        const v4f fA = *(const v4f*)&s_g[1 * 1024 + gbase];
        const v4f fB = *(const v4f*)&s_g[1 * 1024 + gbase + 4];
        const v4f gA = *(const v4f*)&s_g[2 * 1024 + gbase];
        const v4f gB = *(const v4f*)&s_g[2 * 1024 + gbase + 4];
        const v4f oA = *(const v4f*)&s_g[3 * 1024 + gbase];
        const v4f oB = *(const v4f*)&s_g[3 * 1024 + gbase + 4];

        float part = 0.0f;
        v8h hpack;
#pragma unroll
        for (int j = 0; j < 8; ++j) {
            const float iv = (j < 4) ? iA[j] : iB[j - 4];
            const float fv = (j < 4) ? fA[j] : fB[j - 4];
            const float gv = (j < 4) ? gA[j] : gB[j - 4];
            const float ov = (j < 4) ? oA[j] : oB[j - 4];
            const float cc = fmaf(fv, c8[j], iv * gv);
            c8[j] = cc;
            const float hh = ov * fast_tanh(cc);
            part = fmaf(hh, wfc[j], part);
            hpack[j] = (_Float16)hh;
        }
        *(v8h*)&s_h[erow * HSZ + eu0] = hpack;       // h for next step (f16)

        // reduce FC partial across the 8 lanes of this row
        part += __shfl_xor(part, 4, 32);
        part += __shfl_xor(part, 2, 32);
        part += __shfl_xor(part, 1, 32);
        if ((tid & 7) == 0)
            out[(size_t)(bb + erow) * TLEN + t] = part + bfc;

        __syncthreads();   // protect s_h (and s_g reuse) before next step
    }
}

extern "C" void kernel_launch(void* const* d_in, const int* in_sizes, int n_in,
                              void* d_out, int out_size, void* d_ws, size_t ws_size,
                              hipStream_t stream) {
    const float* x    = (const float*)d_in[0];
    const float* W_ih = (const float*)d_in[1];
    const float* W_hh = (const float*)d_in[2];
    const float* b_ih = (const float*)d_in[3];
    const float* b_hh = (const float*)d_in[4];
    const float* W_fc = (const float*)d_in[5];
    const float* b_fc = (const float*)d_in[6];
    float* out = (float*)d_out;

    lstm_persistent<<<dim3(64), dim3(128), 0, stream>>>(
        x, W_ih, W_hh, b_ih, b_hh, W_fc, b_fc, out);
}
